// GearNetFromCoordinates_16381005267596
// MI455X (gfx1250) — compile-verified, matrix operand order checked by reading, not run on hardware
//
#include <hip/hip_runtime.h>
#include <hip/hip_bf16.h>
#include <stdint.h>

#define BB   4
#define LL   2048
#define KNN  10
#define HID  512
#define RRel 7
#define BN_EPS 1e-5f

typedef __bf16 bf16;
typedef __attribute__((ext_vector_type(16))) __bf16 v16bf;
typedef __attribute__((ext_vector_type(8)))  float  v8f;

// ---------------------------------------------------------------- kNN -------
__global__ void knn_kernel(const float* __restrict__ pos, int* __restrict__ nn) {
  int node = blockIdx.x;
  int b = node / LL, i = node % LL;
  const float* P = pos + (size_t)b * LL * 3;
  float xi = P[i * 3 + 0], yi = P[i * 3 + 1], zi = P[i * 3 + 2];
  int lane = threadIdx.x;

  float bd[KNN]; int bi[KNN];
#pragma unroll
  for (int t = 0; t < KNN; ++t) { bd[t] = 3.4e38f; bi[t] = 0; }

  for (int j = lane; j < LL; j += 32) {
    if (j == i) continue;
    float dx = P[j * 3 + 0] - xi, dy = P[j * 3 + 1] - yi, dz = P[j * 3 + 2] - zi;
    float d = dx * dx + dy * dy + dz * dz;   // squared dist: same ordering
    int id = j;
#pragma unroll
    for (int t = 0; t < KNN; ++t) {
      if (d < bd[t]) {
        float td = bd[t]; bd[t] = d;  d = td;
        int   ti = bi[t]; bi[t] = id; id = ti;
      }
    }
  }

  __shared__ float sd[32 * KNN];
  __shared__ int   si[32 * KNN];
#pragma unroll
  for (int t = 0; t < KNN; ++t) { sd[lane * KNN + t] = bd[t]; si[lane * KNN + t] = bi[t]; }
  __syncthreads();

  if (lane == 0) {
    int head[32];
    for (int w = 0; w < 32; ++w) head[w] = 0;
    int* out = nn + (size_t)node * KNN;
    for (int t = 0; t < KNN; ++t) {
      float best = 3.5e38f; int bw = 0;
      for (int w = 0; w < 32; ++w) {
        if (head[w] < KNN) {
          float v = sd[w * KNN + head[w]];
          if (v < best) { best = v; bw = w; }
        }
      }
      out[t] = si[bw * KNN + head[bw]];
      head[bw]++;
    }
  }
}

// ------------------------------------------------- reverse-CSR construction -
__global__ void zero_i32(int* p, int n) {
  int t = blockIdx.x * blockDim.x + threadIdx.x;
  if (t < n) p[t] = 0;
}

__global__ void count_kernel(const int* __restrict__ nn, int* __restrict__ cnt) {
  int e = blockIdx.x * blockDim.x + threadIdx.x;
  if (e >= BB * LL * KNN) return;
  int b = e / (LL * KNN);
  atomicAdd(cnt + b * LL + nn[e], 1);
}

__global__ void scan_kernel(const int* __restrict__ cnt, int* __restrict__ offs) {
  int b = blockIdx.x;
  if (threadIdx.x != 0) return;
  const int* c = cnt + b * LL;
  int* o = offs + b * (LL + 1);
  int s = 0;
  for (int i = 0; i < LL; ++i) { o[i] = s; s += c[i]; }
  o[LL] = s;
}

__global__ void fill_kernel(const int* __restrict__ nn, const int* __restrict__ offs,
                            int* __restrict__ cursor, int* __restrict__ rev) {
  int e = blockIdx.x * blockDim.x + threadIdx.x;
  if (e >= BB * LL * KNN) return;
  int b = e / (LL * KNN);
  int i = (e / KNN) % LL;
  int d = nn[e];
  int p = atomicAdd(cursor + b * LL + d, 1);
  rev[b * LL * KNN + offs[b * (LL + 1) + d] + p] = i;
}

// -------------------------------------------------- per-layer weight prep ---
__global__ void prep_w_kernel(const float* __restrict__ W, const float* __restrict__ S,
                              bf16* __restrict__ WT, int din, int kpad) {
  int t = blockIdx.x * blockDim.x + threadIdx.x;
  if (t >= HID * kpad) return;
  int n = t / kpad, k = t % kpad;
  float v = 0.f;
  if (k < 3 * din)      v = W[n * (RRel * din) + k];        // rel 0..2 blocks
  else if (k < 4 * din) v = S[n * din + (k - 3 * din)];     // self-loop block
  WT[t] = (bf16)v;
}

__global__ void prep_bn_kernel(const float* bW, const float* bS, const float* gamma,
                               const float* beta, const float* mean, const float* var,
                               float* scale, float* shift) {
  int n = blockIdx.x * blockDim.x + threadIdx.x;
  if (n >= HID) return;
  float a = gamma[n] * rsqrtf(var[n] + BN_EPS);
  scale[n] = a;
  shift[n] = a * (bW[n] + bS[n] - mean[n]) + beta[n];
}

// --------------------------------------------------------- aggregation ------
__global__ void agg_kernel(const float* __restrict__ x, const int* __restrict__ offs,
                           const int* __restrict__ rev, bf16* __restrict__ M,
                           int din, int kpad) {
  int node = blockIdx.x;
  int b = node / LL, d = node % LL;
  const float* X = x + (size_t)b * LL * din;
  bf16* Mo = M + (size_t)node * kpad;
  const int* O = offs + b * (LL + 1);
  int e0 = O[d], e1 = O[d + 1];
  const int* rv = rev + (size_t)b * LL * KNN;

  for (int c = threadIdx.x; c < din; c += blockDim.x) {
    float s0 = 0.f, s1 = 0.f, s2 = 0.f;
#pragma unroll
    for (int o = 1; o <= 3; ++o) {
      if (d + o < LL) s0 += X[(d + o) * din + c];   // rel 0 (negative offsets)
      if (d - o >= 0) s1 += X[(d - o) * din + c];   // rel 1 (positive offsets)
    }
    for (int e = e0; e < e1; ++e) s2 += X[rv[e] * din + c];  // rel 2 (kNN)
    Mo[0 * din + c] = (bf16)s0;
    Mo[1 * din + c] = (bf16)s1;
    Mo[2 * din + c] = (bf16)s2;
    Mo[3 * din + c] = (bf16)X[d * din + c];
  }
  for (int k = 4 * din + threadIdx.x; k < kpad; k += blockDim.x) Mo[k] = (bf16)0.f;
}

// ------------------------------------------------------------- WMMA GEMM ----
// C[b] = M[b] (LLxK bf16) x WT^T (KxHID) with fused BN/ReLU/residual.
// Block tile 128x128x32, 4 waves (2x2), 64x64 per wave (4x4 accum subtiles).
// Double-buffered LDS filled by global_load_async_to_lds_b128 (ASYNCcnt).
#define BM 128
#define BN 128
#define BK 32
#define LDT 40   // padded LDS row stride (bf16): 80 B, 16B-aligned rows

struct Frag32 { uint4 lo, hi; };

// Issue 8 async 16B loads per thread: 128x32 bf16 A tile + 128x32 B tile.
__device__ __forceinline__ void stage_async(const bf16* __restrict__ Ag,
                                            const bf16* __restrict__ Bg,
                                            bf16* AtBuf, bf16* BtBuf,
                                            int tid, int kdim) {
  int r0 = tid >> 2;          // 0..31
  int cc = (tid & 3) * 8;     // 0,8,16,24
#pragma unroll
  for (int it = 0; it < 4; ++it) {
    int row = r0 + it * 32;
    const void* ga = Ag + (size_t)row * kdim + cc;
    uint32_t la = (uint32_t)(uintptr_t)(AtBuf + row * LDT + cc);
    asm volatile("global_load_async_to_lds_b128 %0, %1, off"
                 :: "v"(la), "v"(ga) : "memory");
    const void* gb = Bg + (size_t)row * kdim + cc;
    uint32_t lb = (uint32_t)(uintptr_t)(BtBuf + row * LDT + cc);
    asm volatile("global_load_async_to_lds_b128 %0, %1, off"
                 :: "v"(lb), "v"(gb) : "memory");
  }
}

__global__ __launch_bounds__(128) void gemm_kernel(
    const bf16* __restrict__ Mbuf, const bf16* __restrict__ WT,
    const float* __restrict__ scale, const float* __restrict__ shift,
    const float* __restrict__ xprev, float* __restrict__ out,
    int kdim, int residual) {
  __shared__ __align__(16) bf16 At[2][BM * LDT];
  __shared__ __align__(16) bf16 Bt[2][BN * LDT];

  int b  = blockIdx.z;
  int m0 = blockIdx.x * BM;
  int n0 = blockIdx.y * BN;
  const bf16* A  = Mbuf + (size_t)b * LL * kdim + (size_t)m0 * kdim;
  const bf16* Bw = WT + (size_t)n0 * kdim;

  int tid  = threadIdx.x;
  int w    = tid >> 5, lane = tid & 31;
  int wm   = (w & 1) * 64;   // wave M offset in block tile
  int wn   = (w >> 1) * 64;  // wave N offset in block tile
  int kh   = lane >> 4;      // K-half per documented bf16 A/B lane layout
  int mr   = lane & 15;      // row (A) / col (B) within 16

  v8f acc[4][4];
#pragma unroll
  for (int im = 0; im < 4; ++im)
#pragma unroll
    for (int in = 0; in < 4; ++in)
#pragma unroll
      for (int r = 0; r < 8; ++r) acc[im][in][r] = 0.f;

  int nK = kdim / BK;
  // Prologue: async-stage tile 0 into buffer 0.
  stage_async(A, Bw, At[0], Bt[0], tid, kdim);

  for (int kt = 0; kt < nK; ++kt) {
    int cur = kt & 1;
    if (kt + 1 < nK) {
      // Overlap: issue next tile into the other buffer, then wait for the
      // current tile's 8 in-order async loads (<=8 outstanding left).
      stage_async(A + (kt + 1) * BK, Bw + (kt + 1) * BK,
                  At[cur ^ 1], Bt[cur ^ 1], tid, kdim);
      asm volatile("s_wait_asynccnt 0x8" ::: "memory");
    } else {
      asm volatile("s_wait_asynccnt 0x0" ::: "memory");
    }
    __syncthreads();

    v16bf af[4], bfr[4];
#pragma unroll
    for (int im = 0; im < 4; ++im) {
      const bf16* p = &At[cur][(wm + im * 16 + mr) * LDT + kh * 8];
      Frag32 f;
      f.lo = *reinterpret_cast<const uint4*>(p);        // K = kh*8 .. +7
      f.hi = *reinterpret_cast<const uint4*>(p + 16);   // K = 16 + kh*8 .. +7
      af[im] = __builtin_bit_cast(v16bf, f);
    }
#pragma unroll
    for (int in = 0; in < 4; ++in) {
      const bf16* p = &Bt[cur][(wn + in * 16 + mr) * LDT + kh * 8];
      Frag32 f;
      f.lo = *reinterpret_cast<const uint4*>(p);
      f.hi = *reinterpret_cast<const uint4*>(p + 16);
      bfr[in] = __builtin_bit_cast(v16bf, f);
    }

#pragma unroll
    for (int im = 0; im < 4; ++im)
#pragma unroll
      for (int in = 0; in < 4; ++in)
        acc[im][in] = __builtin_amdgcn_wmma_f32_16x16x32_bf16(
            false, af[im], false, bfr[in], (short)0, acc[im][in], false, false);
    __syncthreads();   // protect buffer reuse before next iteration's staging
  }

  // Epilogue per documented 16x16 f32 C layout.
#pragma unroll
  for (int im = 0; im < 4; ++im) {
#pragma unroll
    for (int in = 0; in < 4; ++in) {
      int col = n0 + wn + in * 16 + mr;
      float sc = scale[col], sh = shift[col];
#pragma unroll
      for (int r = 0; r < 8; ++r) {
        int row = m0 + wm + im * 16 + kh * 8 + r;
        float v = acc[im][in][r] * sc + sh;
        v = v > 0.f ? v : 0.f;                       // ReLU
        size_t oidx = ((size_t)b * LL + row) * HID + col;
        if (residual) v += xprev[oidx];              // short_cut (in-place safe)
        out[oidx] = v;
      }
    }
  }
}

// ----------------------------------------------------------- host driver ----
extern "C" void kernel_launch(void* const* d_in, const int* in_sizes, int n_in,
                              void* d_out, int out_size, void* d_ws, size_t ws_size,
                              hipStream_t stream) {
  (void)in_sizes; (void)n_in; (void)out_size; (void)ws_size;
  const float* ca = (const float*)d_in[1];  // ca_coords [B, L, 3]

  char* ws = (char*)d_ws;
  size_t off = 0;
  auto carve = [&](size_t bytes) -> void* {
    void* p = ws + off;
    off = (off + bytes + 255) & ~(size_t)255;
    return p;
  };
  int*   nn     = (int*)  carve((size_t)BB * LL * KNN * 4);
  int*   cnt    = (int*)  carve((size_t)BB * LL * 4);
  int*   offs   = (int*)  carve((size_t)BB * (LL + 1) * 4);
  int*   cursor = (int*)  carve((size_t)BB * LL * 4);
  int*   rev    = (int*)  carve((size_t)BB * LL * KNN * 4);
  bf16*  WT     = (bf16*) carve((size_t)HID * 2048 * 2);
  float* scale  = (float*)carve((size_t)HID * 4);
  float* shift  = (float*)carve((size_t)HID * 4);
  bf16*  Mbuf   = (bf16*) carve((size_t)BB * LL * 2048 * 2);
  float* xbuf   = (float*)carve((size_t)BB * LL * HID * 4);

  // Graph construction (shared by all 4 layers).
  knn_kernel<<<BB * LL, 32, 0, stream>>>(ca, nn);
  zero_i32<<<(BB * LL + 255) / 256, 256, 0, stream>>>(cnt, BB * LL);
  count_kernel<<<(BB * LL * KNN + 255) / 256, 256, 0, stream>>>(nn, cnt);
  scan_kernel<<<BB, 32, 0, stream>>>(cnt, offs);
  zero_i32<<<(BB * LL + 255) / 256, 256, 0, stream>>>(cursor, BB * LL);
  fill_kernel<<<(BB * LL * KNN + 255) / 256, 256, 0, stream>>>(nn, offs, cursor, rev);

  // Params flatten: 3 coord arrays, then per-layer dicts in sorted-key order:
  // S, W, bS, bW, beta, gamma, mean, var.
  for (int layer = 0; layer < 4; ++layer) {
    int base = 3 + layer * 8;
    const float* S     = (const float*)d_in[base + 0];
    const float* W     = (const float*)d_in[base + 1];
    const float* bS    = (const float*)d_in[base + 2];
    const float* bW    = (const float*)d_in[base + 3];
    const float* beta  = (const float*)d_in[base + 4];
    const float* gamma = (const float*)d_in[base + 5];
    const float* mean  = (const float*)d_in[base + 6];
    const float* var   = (const float*)d_in[base + 7];

    int din  = (layer == 0) ? 3  : HID;
    int kpad = (layer == 0) ? 32 : 4 * HID;   // 32 (padded) or 2048

    prep_w_kernel<<<(HID * kpad + 255) / 256, 256, 0, stream>>>(W, S, WT, din, kpad);
    prep_bn_kernel<<<(HID + 255) / 256, 256, 0, stream>>>(bW, bS, gamma, beta, mean, var,
                                                          scale, shift);

    const float* xin = (layer == 0) ? ca : xbuf;
    float* xout = (layer == 3) ? (float*)d_out : xbuf;  // in-place residual OK

    agg_kernel<<<BB * LL, 256, 0, stream>>>(xin, offs, rev, Mbuf, din, kpad);
    gemm_kernel<<<dim3(LL / BM, HID / BN, BB), 128, 0, stream>>>(
        Mbuf, WT, scale, shift, xin, xout, kpad, layer > 0 ? 1 : 0);
  }
}
